// IRNetLayer_49392123904433
// MI455X (gfx1250) — compile-verified, matrix operand order checked by reading, not run on hardware
//
#include <hip/hip_runtime.h>

// ---------------- problem constants ----------------
constexpr int Bc = 8;        // batch
constexpr int Sc = 1024;     // seq
constexpr int Hc = 1024;     // hidden
constexpr int NHc = 16;      // heads
constexpr int HDc = 64;      // head dim
constexpr int FFc = 4096;    // ff dim

typedef __attribute__((ext_vector_type(16))) __bf16        v16bf;
typedef __attribute__((ext_vector_type(8)))  float         v8f;
typedef __attribute__((ext_vector_type(4)))  unsigned int  u32x4;

union Frag {
    v16bf v;
    u32x4 u[2];
};
union Pack8 {
    __bf16 h[8];
    u32x4  u;
};

__device__ __forceinline__ v8f wmma_bf16(const Frag& a, const Frag& b, v8f c) {
    return __builtin_amdgcn_wmma_f32_16x16x32_bf16(
        false, a.v, false, b.v, (short)0, c, false, false);
}

// ---------------- f32 -> bf16 cast (elementwise) ----------------
__global__ void cast_f32_to_bf16(const float* __restrict__ in,
                                 __bf16* __restrict__ out, int n4) {
    int i = blockIdx.x * blockDim.x + threadIdx.x;
    if (i < n4) {
        const float4 v = *(const float4*)(in + (size_t)i * 4);
        size_t o = (size_t)i * 4;
        out[o + 0] = (__bf16)v.x;
        out[o + 1] = (__bf16)v.y;
        out[o + 2] = (__bf16)v.z;
        out[o + 3] = (__bf16)v.w;
    }
}

// ---------------- f32 [K][N] -> bf16 [N][K] transpose-cast ----------------
__global__ __launch_bounds__(256) void cast_transpose_f32_bf16(
    const float* __restrict__ in, __bf16* __restrict__ out, int K, int N)
{
    __shared__ float tile[32][33];
    const int nT = blockIdx.x * 32;
    const int kT = blockIdx.y * 32;
    const int tx = threadIdx.x & 31;
    const int ty = threadIdx.x >> 5;
#pragma unroll
    for (int j = 0; j < 4; ++j)
        tile[ty + j * 8][tx] = in[(size_t)(kT + ty + j * 8) * N + nT + tx];
    __syncthreads();
#pragma unroll
    for (int j = 0; j < 4; ++j)
        out[(size_t)(nT + ty + j * 8) * K + kT + tx] = (__bf16)tile[tx][ty + j * 8];
}

// ---------------- mask pack: bool [B][S][S] -> u16 bits over q: [B][key][S/16] ----
__global__ __launch_bounds__(256) void pack_mask_T(
    const unsigned char* __restrict__ mask, unsigned short* __restrict__ mt)
{
    __shared__ unsigned char t[16][64];
    const int b  = blockIdx.z;
    const int q0 = blockIdx.y * 16;
    const int k0 = blockIdx.x * 64;
    const int tx = threadIdx.x & 63;   // key offset
    const int ty = threadIdx.x >> 6;   // 0..3
#pragma unroll
    for (int j = 0; j < 4; ++j)
        t[ty + 4 * j][tx] = mask[((size_t)b * Sc + q0 + ty + 4 * j) * Sc + k0 + tx];
    __syncthreads();
    if (threadIdx.x < 64) {
        unsigned int w = 0;
#pragma unroll
        for (int j = 0; j < 16; ++j) w |= (t[j][threadIdx.x] ? 1u : 0u) << j;
        mt[((size_t)b * Sc + k0 + threadIdx.x) * (Sc / 16) + blockIdx.y] =
            (unsigned short)w;
    }
}

// ---------------- generic bf16 WMMA GEMM (B pre-transposed) ----------------
#define BM 128
#define BN 128
#define BK 32

template <int RELU, int HAS_RESID, int OUT_F, int OUT_H, int QKV>
__global__ __launch_bounds__(256) void gemm_bf16_wmma(
    const __bf16* __restrict__ A, const __bf16* __restrict__ Bt,
    const float* __restrict__ bias, const float* __restrict__ resid,
    float* __restrict__ outF, __bf16* __restrict__ outH,
    __bf16* __restrict__ vOutT,
    int M, int N, int K)
{
    __shared__ __bf16 sA[2][BM][BK];
    __shared__ __bf16 sB[2][BN][BK];

    const int tid  = threadIdx.x;
    const int wid  = tid >> 5;
    const int lane = tid & 31;
    const int half = lane >> 4;
    const int l16  = lane & 15;
    const int wm = wid & 3;
    const int wn = wid >> 2;

    const size_t rowBase = (size_t)blockIdx.y * BM;
    const size_t colBase = (size_t)blockIdx.x * BN;

    const int sr0 = tid >> 2,         sc0 = (tid & 3) * 8;
    const int sr1 = (tid + 256) >> 2, sc1 = ((tid + 256) & 3) * 8;

    v8f zero = {};
    v8f acc[2][4];
#pragma unroll
    for (int i = 0; i < 2; ++i)
#pragma unroll
        for (int j = 0; j < 4; ++j) acc[i][j] = zero;

    u32x4 pa0, pa1, pb0, pb1;
    pa0 = *(const u32x4*)(A  + (rowBase + sr0) * (size_t)K + sc0);
    pa1 = *(const u32x4*)(A  + (rowBase + sr1) * (size_t)K + sc1);
    pb0 = *(const u32x4*)(Bt + (colBase + sr0) * (size_t)K + sc0);
    pb1 = *(const u32x4*)(Bt + (colBase + sr1) * (size_t)K + sc1);
    *(u32x4*)&sA[0][sr0][sc0] = pa0;
    *(u32x4*)&sA[0][sr1][sc1] = pa1;
    *(u32x4*)&sB[0][sr0][sc0] = pb0;
    *(u32x4*)&sB[0][sr1][sc1] = pb1;
    __syncthreads();

    int cur = 0;
    for (int kb = 0; kb < K; kb += BK) {
        const int nxt = kb + BK;
        const bool more = nxt < K;
        if (more) {
            pa0 = *(const u32x4*)(A  + (rowBase + sr0) * (size_t)K + nxt + sc0);
            pa1 = *(const u32x4*)(A  + (rowBase + sr1) * (size_t)K + nxt + sc1);
            pb0 = *(const u32x4*)(Bt + (colBase + sr0) * (size_t)K + nxt + sc0);
            pb1 = *(const u32x4*)(Bt + (colBase + sr1) * (size_t)K + nxt + sc1);
        }

        Frag afr[2], bfr[4];
#pragma unroll
        for (int tm = 0; tm < 2; ++tm) {
            int m = wm * 32 + tm * 16 + l16;
            afr[tm].u[0] = *(const u32x4*)&sA[cur][m][half * 8];
            afr[tm].u[1] = *(const u32x4*)&sA[cur][m][16 + half * 8];
        }
#pragma unroll
        for (int tn = 0; tn < 4; ++tn) {
            int n = wn * 64 + tn * 16 + l16;
            bfr[tn].u[0] = *(const u32x4*)&sB[cur][n][half * 8];
            bfr[tn].u[1] = *(const u32x4*)&sB[cur][n][16 + half * 8];
        }
#pragma unroll
        for (int tm = 0; tm < 2; ++tm)
#pragma unroll
            for (int tn = 0; tn < 4; ++tn)
                acc[tm][tn] = wmma_bf16(afr[tm], bfr[tn], acc[tm][tn]);

        if (more) {
            int nb = cur ^ 1;
            *(u32x4*)&sA[nb][sr0][sc0] = pa0;
            *(u32x4*)&sA[nb][sr1][sc1] = pa1;
            *(u32x4*)&sB[nb][sr0][sc0] = pb0;
            *(u32x4*)&sB[nb][sr1][sc1] = pb1;
            __syncthreads();
        }
        cur ^= 1;
    }

    // ---- epilogue ----
    if (QKV && colBase >= (size_t)(2 * Hc)) {
        // V region: write transposed vT[b][h][hd][S], 16B packed per tile
        const size_t bb = rowBase >> 10;                 // / Sc
        const size_t sRow = (rowBase & (size_t)(Sc - 1)) + wm * 32;
#pragma unroll
        for (int tm = 0; tm < 2; ++tm)
#pragma unroll
            for (int tn = 0; tn < 4; ++tn) {
                size_t col = colBase + wn * 64 + tn * 16 + l16;
                float bv = bias[col];
                int c2 = (int)col - 2 * Hc;              // 0..1023
                int hh = c2 >> 6, hd = c2 & 63;
                Pack8 pk;
#pragma unroll
                for (int r = 0; r < 8; ++r)
                    pk.h[r] = (__bf16)(acc[tm][tn][r] + bv);
                size_t s0 = sRow + tm * 16 + half * 8;   // 8 consecutive tokens
                *(u32x4*)&vOutT[(((bb * NHc + hh) * HDc + hd) * Sc) + s0] = pk.u;
            }
        return;
    }

    const size_t ldH = QKV ? (size_t)(2 * Hc) : (size_t)N;
#pragma unroll
    for (int tm = 0; tm < 2; ++tm)
#pragma unroll
        for (int tn = 0; tn < 4; ++tn) {
            size_t col = colBase + wn * 64 + tn * 16 + l16;
            float bv = bias[col];
#pragma unroll
            for (int r = 0; r < 8; ++r) {
                size_t row = rowBase + wm * 32 + tm * 16 + r + half * 8;
                float v = acc[tm][tn][r] + bv;
                if (RELU) v = fmaxf(v, 0.0f);
                if (HAS_RESID) v += resid[row * (size_t)N + col];
                if (OUT_F) outF[row * (size_t)N + col] = v;
                if (OUT_H) outH[row * ldH + col] = (__bf16)v;
            }
        }
}

// ---------------- flash attention (barrier-free, u32-offset hot loop) --------
// grid: (S/64, NH, B), 128 threads = 4 waves, wave owns 16 q rows.
// 32-bit element offsets off the kernarg bases keep the loads on the GLOBAL
// path (saddr + 32-bit voffset) instead of FLAT, so they only touch LOADcnt.
__global__ __launch_bounds__(128) void flash_attn_bf16(
    const __bf16* __restrict__ qk,          // [B*S, 2*H] bf16 (q | k)
    const __bf16* __restrict__ vT,          // [B][NH][HD][S] bf16
    const unsigned short* __restrict__ mt,  // [B][key][S/16] q-bitmask
    __bf16* __restrict__ out)               // [B*S, H] bf16 (head-major cols)
{
    __shared__ __bf16 sP[4][16][32];        // per-wave P staging [m][k]

    const int tid  = threadIdx.x;
    const int wid  = tid >> 5;
    const int lane = tid & 31;
    const int half = lane >> 4;
    const int l16  = lane & 15;
    const int b = blockIdx.z, h = blockIdx.y;
    const int qbase = blockIdx.x * 64 + wid * 16;
    const int qword = qbase >> 4;           // index into per-key q-bitmask
    const float scale = 0.125f;             // 1/sqrt(64)
    const int h8 = half * 8;

    // Q fragments (A-matrix 16 x 64 -> 2 frags of K=32), held in registers
    Frag qa[2];
    {
        int q = qbase + l16;
        const __bf16* qrow = qk + (size_t)(b * Sc + q) * (2 * Hc) + h * HDc;
#pragma unroll
        for (int f = 0; f < 2; ++f) {
            qa[f].u[0] = *(const u32x4*)(qrow + f * 32 + h8);
            qa[f].u[1] = *(const u32x4*)(qrow + f * 32 + 16 + h8);
        }
    }

    // strength-reduced 32-bit element offsets (advanced by constants)
    unsigned koff[2], moff[2], voff[4];
#pragma unroll
    for (int t = 0; t < 2; ++t) {
        unsigned key = (unsigned)(t * 16 + l16);
        koff[t] = (unsigned)(b * Sc + key) * (unsigned)(2 * Hc)
                  + (unsigned)(Hc + h * HDc);
        moff[t] = (unsigned)(b * Sc + key) * (unsigned)(Sc / 16) + (unsigned)qword;
    }
#pragma unroll
    for (int t = 0; t < 4; ++t)
        voff[t] = (unsigned)(b * NHc + h) * (unsigned)(HDc * Sc)
                  + (unsigned)(t * 16 + l16) * (unsigned)Sc;
    const unsigned kStep = 32u * (unsigned)(2 * Hc);
    const unsigned mStep = 32u * (unsigned)(Sc / 16);

    v8f zero = {};
    v8f o[4];
    float mrow[8], lrow[8];
#pragma unroll
    for (int t = 0; t < 4; ++t) o[t] = zero;
#pragma unroll
    for (int r = 0; r < 8; ++r) { mrow[r] = -1e30f; lrow[r] = 0.0f; }

    for (int kb = 0; kb < Sc; kb += 32) {
        // scores S = Q * K^T : 2 n-tiles of 16 keys, contraction 64 (2 wmma)
        v8f s[2];
        unsigned int mbits[2];
#pragma unroll
        for (int t = 0; t < 2; ++t) {
            Frag kf0, kf1;
            kf0.u[0] = *(const u32x4*)(qk + koff[t] + h8);
            kf0.u[1] = *(const u32x4*)(qk + koff[t] + 16 + h8);
            kf1.u[0] = *(const u32x4*)(qk + koff[t] + 32 + h8);
            kf1.u[1] = *(const u32x4*)(qk + koff[t] + 48 + h8);
            mbits[t] = mt[moff[t]];
            v8f a = zero;
            a = wmma_bf16(qa[0], kf0, a);
            a = wmma_bf16(qa[1], kf1, a);
            s[t] = a;
            koff[t] += kStep;
            moff[t] += mStep;
        }

        // scale + mask via pre-packed q-bitmask (bit index == r + 8*half)
#pragma unroll
        for (int t = 0; t < 2; ++t)
#pragma unroll
            for (int r = 0; r < 8; ++r) {
                float v = s[t][r] * scale;
                if ((mbits[t] >> (r + 8 * half)) & 1u) v = -1e20f;
                s[t][r] = v;
            }

        // online softmax: row (r + 8*half), columns live in a 16-lane group
        float alpha[8];
#pragma unroll
        for (int r = 0; r < 8; ++r) {
            float v = fmaxf(s[0][r], s[1][r]);
            v = fmaxf(v, __shfl_xor(v, 8, 16));
            v = fmaxf(v, __shfl_xor(v, 4, 16));
            v = fmaxf(v, __shfl_xor(v, 2, 16));
            v = fmaxf(v, __shfl_xor(v, 1, 16));
            float mn = fmaxf(mrow[r], v);
            alpha[r] = __expf(mrow[r] - mn);
            mrow[r] = mn;
            float p0 = __expf(s[0][r] - mn);
            float p1 = __expf(s[1][r] - mn);
            s[0][r] = p0; s[1][r] = p1;
            float rs = p0 + p1;
            rs += __shfl_xor(rs, 8, 16);
            rs += __shfl_xor(rs, 4, 16);
            rs += __shfl_xor(rs, 2, 16);
            rs += __shfl_xor(rs, 1, 16);
            lrow[r] = lrow[r] * alpha[r] + rs;
        }

        // rescale running O, stage P (bf16) per-wave as [m][k]
#pragma unroll
        for (int t = 0; t < 4; ++t)
#pragma unroll
            for (int r = 0; r < 8; ++r) o[t][r] *= alpha[r];
#pragma unroll
        for (int t = 0; t < 2; ++t)
#pragma unroll
            for (int r = 0; r < 8; ++r)
                sP[wid][r + half * 8][t * 16 + l16] = (__bf16)s[t][r];

        // O += P * V : V fragments straight from global (k-contiguous rows)
        Frag pf;
        pf.u[0] = *(const u32x4*)&sP[wid][l16][h8];
        pf.u[1] = *(const u32x4*)&sP[wid][l16][16 + h8];
#pragma unroll
        for (int t = 0; t < 4; ++t) {
            Frag vf;
            vf.u[0] = *(const u32x4*)(vT + voff[t] + h8);
            vf.u[1] = *(const u32x4*)(vT + voff[t] + 16 + h8);
            o[t] = wmma_bf16(pf, vf, o[t]);
            voff[t] += 32u;
        }
    }

    // normalize and write out: col = h*64 + hd
#pragma unroll
    for (int t = 0; t < 4; ++t)
#pragma unroll
        for (int r = 0; r < 8; ++r) {
            int q = qbase + r + half * 8;
            int col = h * HDc + t * 16 + l16;
            out[(size_t)(b * Sc + q) * Hc + col] = (__bf16)(o[t][r] / lrow[r]);
        }
}

// ---------------- LayerNorm (block per row, H=1024, 256 threads) ----------------
__global__ __launch_bounds__(256) void layernorm_k(
    const float* __restrict__ x, const float* __restrict__ g,
    const float* __restrict__ beta,
    float* __restrict__ yF, __bf16* __restrict__ yH)
{
    __shared__ float red[256];
    const size_t row = blockIdx.x;
    const float* xr = x + row * Hc;
    float local[4];
    float s = 0.0f;
#pragma unroll
    for (int i = 0; i < 4; ++i) { local[i] = xr[threadIdx.x + i * 256]; s += local[i]; }
    red[threadIdx.x] = s;
    __syncthreads();
    for (int st = 128; st > 0; st >>= 1) {
        if ((int)threadIdx.x < st) red[threadIdx.x] += red[threadIdx.x + st];
        __syncthreads();
    }
    float mu = red[0] * (1.0f / Hc);
    __syncthreads();
    float vs = 0.0f;
#pragma unroll
    for (int i = 0; i < 4; ++i) { float d = local[i] - mu; vs += d * d; }
    red[threadIdx.x] = vs;
    __syncthreads();
    for (int st = 128; st > 0; st >>= 1) {
        if ((int)threadIdx.x < st) red[threadIdx.x] += red[threadIdx.x + st];
        __syncthreads();
    }
    float rstd = rsqrtf(red[0] * (1.0f / Hc) + 1e-5f);
#pragma unroll
    for (int i = 0; i < 4; ++i) {
        int c = threadIdx.x + i * 256;
        float v = (local[i] - mu) * rstd * g[c] + beta[c];
        if (yF) yF[row * Hc + c] = v;
        if (yH) yH[row * Hc + c] = (__bf16)v;
    }
}

// ---------------- launcher ----------------
extern "C" void kernel_launch(void* const* d_in, const int* in_sizes, int n_in,
                              void* d_out, int out_size, void* d_ws, size_t ws_size,
                              hipStream_t stream) {
    const float*         x    = (const float*)d_in[0];
    const unsigned char* mask = (const unsigned char*)d_in[1];
    const float* Wqkv = (const float*)d_in[2];
    const float* bqkv = (const float*)d_in[3];
    const float* Wo   = (const float*)d_in[4];
    const float* bo   = (const float*)d_in[5];
    const float* g1   = (const float*)d_in[6];
    const float* b1   = (const float*)d_in[7];
    const float* W1   = (const float*)d_in[8];
    const float* bf1  = (const float*)d_in[9];
    const float* W2   = (const float*)d_in[10];
    const float* bf2  = (const float*)d_in[11];
    const float* g2   = (const float*)d_in[12];
    const float* b2   = (const float*)d_in[13];
    float* out = (float*)d_out;

    const size_t BS = (size_t)Bc * Sc;   // 8192 rows

    char* p = (char*)d_ws;
    auto alloc = [&](size_t bytes) -> void* {
        void* r = (void*)p;
        p += (bytes + 255) & ~(size_t)255;
        return r;
    };
    __bf16* x_h     = (__bf16*)alloc(BS * Hc * sizeof(__bf16));
    __bf16* WqkvT_h = (__bf16*)alloc((size_t)Hc * 3 * Hc * sizeof(__bf16));
    __bf16* WoT_h   = (__bf16*)alloc((size_t)Hc * Hc * sizeof(__bf16));
    __bf16* W1T_h   = (__bf16*)alloc((size_t)Hc * FFc * sizeof(__bf16));
    __bf16* W2T_h   = (__bf16*)alloc((size_t)FFc * Hc * sizeof(__bf16));
    __bf16* qk_h    = (__bf16*)alloc(BS * 2 * Hc * sizeof(__bf16));
    __bf16* vT_h    = (__bf16*)alloc(BS * Hc * sizeof(__bf16));
    unsigned short* maskT =
        (unsigned short*)alloc((size_t)Bc * Sc * (Sc / 16) * sizeof(unsigned short));
    __bf16* attn_h  = (__bf16*)alloc(BS * Hc * sizeof(__bf16));
    float*  ln1_in  = (float*)alloc(BS * Hc * sizeof(float));
    float*  y_f     = (float*)alloc(BS * Hc * sizeof(float));
    __bf16* y_h     = (__bf16*)alloc(BS * Hc * sizeof(__bf16));
    __bf16* ff1_h   = (__bf16*)alloc(BS * FFc * sizeof(__bf16));
    float*  ln2_in  = (float*)alloc(BS * Hc * sizeof(float));

    {
        int n4 = (int)(BS * Hc / 4);
        cast_f32_to_bf16<<<(n4 + 255) / 256, 256, 0, stream>>>(x, x_h, n4);
    }
    auto castT = [&](const float* src, __bf16* dst, int K, int N) {
        dim3 g(N / 32, K / 32);
        cast_transpose_f32_bf16<<<g, 256, 0, stream>>>(src, dst, K, N);
    };
    castT(Wqkv, WqkvT_h, Hc, 3 * Hc);
    castT(Wo,   WoT_h,   Hc, Hc);
    castT(W1,   W1T_h,   Hc, FFc);
    castT(W2,   W2T_h,   FFc, Hc);
    {
        dim3 g(Sc / 64, Sc / 16, Bc);
        pack_mask_T<<<g, 256, 0, stream>>>(mask, maskT);
    }

    // QKV projection (q,k -> qk buffer; v -> transposed vT)
    {
        dim3 g(3 * Hc / BN, BS / BM);
        gemm_bf16_wmma<0, 0, 0, 1, 1><<<g, 256, 0, stream>>>(
            x_h, WqkvT_h, bqkv, nullptr, nullptr, qk_h, vT_h,
            (int)BS, 3 * Hc, Hc);
    }
    // Flash attention
    {
        dim3 g(Sc / 64, NHc, Bc);
        flash_attn_bf16<<<g, 128, 0, stream>>>(qk_h, vT_h, maskT, attn_h);
    }
    // O projection + residual(x) -> ln1 input (f32)
    {
        dim3 g(Hc / BN, BS / BM);
        gemm_bf16_wmma<0, 1, 1, 0, 0><<<g, 256, 0, stream>>>(
            attn_h, WoT_h, bo, x, ln1_in, nullptr, nullptr, (int)BS, Hc, Hc);
    }
    // LN1 -> y (f32 for residual, bf16 for FF1)
    layernorm_k<<<(int)BS, 256, 0, stream>>>(ln1_in, g1, b1, y_f, y_h);
    // FF1 + ReLU -> bf16
    {
        dim3 g(FFc / BN, BS / BM);
        gemm_bf16_wmma<1, 0, 0, 1, 0><<<g, 256, 0, stream>>>(
            y_h, W1T_h, bf1, nullptr, nullptr, ff1_h, nullptr, (int)BS, FFc, Hc);
    }
    // FF2 + residual(y) -> ln2 input (f32)
    {
        dim3 g(Hc / BN, BS / BM);
        gemm_bf16_wmma<0, 1, 1, 0, 0><<<g, 256, 0, stream>>>(
            ff1_h, W2T_h, bf2, y_f, ln2_in, nullptr, nullptr, (int)BS, Hc, FFc);
    }
    // LN2 -> final output (f32)
    layernorm_k<<<(int)BS, 256, 0, stream>>>(ln2_in, g2, b2, out, nullptr);
}
